// TCNEstimator_36885179138356
// MI455X (gfx1250) — matrix operand order repaired and checked by reading, not compiled, not run on hardware
//
#include <hip/hip_runtime.h>
#include <hip/hip_bf16.h>

#define BATCH 8
#define CIN   257
#define TLEN  2000
#define TPAD  2048
#define BND   256
#define HID   512
#define NBLK  24
#define CINP  288   // CIN padded to multiple of 32 (K dim)
#define COUTP 272   // CIN padded to multiple of 16 (M dim, final GEMM)

typedef __attribute__((ext_vector_type(16))) __bf16 v16bf;
typedef __attribute__((ext_vector_type(8)))  __bf16 v8bf;
typedef __attribute__((ext_vector_type(8)))  float  v8f;

// ---------------- weight convert (fp32 -> bf16, with zero padding) ----------
__global__ __launch_bounds__(256)
void convert_pad_kernel(const float* __restrict__ src, __bf16* __restrict__ dst,
                        int Rdst, int Cdst, int Rsrc, int Csrc) {
  long i = (long)blockIdx.x * blockDim.x + threadIdx.x;
  long n = (long)Rdst * Cdst;
  if (i >= n) return;
  int r = (int)(i / Cdst), c = (int)(i % Cdst);
  float v = (r < Rsrc && c < Csrc) ? src[(long)r * Csrc + c] : 0.f;
  dst[i] = (__bf16)v;
}

__global__ __launch_bounds__(256)
void zero_f32_kernel(float* __restrict__ p, long n) {
  long i = (long)blockIdx.x * blockDim.x + threadIdx.x;
  if (i < n) p[i] = 0.f;
}

// ---------------- cumulative layer norm pieces ------------------------------
// per-(b,t) channel sum and sum-of-squares; coalesced along t
__global__ __launch_bounds__(256)
void colstats_kernel(const float* __restrict__ a, float* __restrict__ s,
                     float* __restrict__ q, int C) {
  int t = blockIdx.x * blockDim.x + threadIdx.x;
  int b = blockIdx.y;
  if (t >= TLEN) return;
  const float* p = a + (long)b * C * TLEN + t;
  float ss = 0.f, qq = 0.f;
  for (int c = 0; c < C; ++c) { float v = p[(long)c * TLEN]; ss += v; qq += v * v; }
  s[b * TLEN + t] = ss; q[b * TLEN + t] = qq;
}

// per-batch inclusive prefix scan over T, emits mean / inverse-std
__global__ __launch_bounds__(256)
void scan_kernel(const float* __restrict__ s, const float* __restrict__ q,
                 float* __restrict__ mean, float* __restrict__ istd, int C) {
  int b = blockIdx.x, tid = threadIdx.x;
  __shared__ float ls[256], lq[256];
  float cs = 0.f, cq = 0.f;
  for (int base = 0; base < TLEN; base += 256) {
    int t = base + tid;
    ls[tid] = (t < TLEN) ? s[b * TLEN + t] : 0.f;
    lq[tid] = (t < TLEN) ? q[b * TLEN + t] : 0.f;
    __syncthreads();
    for (int off = 1; off < 256; off <<= 1) {
      float as = (tid >= off) ? ls[tid - off] : 0.f;
      float aq = (tid >= off) ? lq[tid - off] : 0.f;
      __syncthreads();
      ls[tid] += as; lq[tid] += aq;
      __syncthreads();
    }
    if (t < TLEN) {
      float ts = cs + ls[tid], tq = cq + lq[tid];
      float cnt = (float)C * (float)(t + 1);
      float m = ts / cnt;
      float var = (tq - 2.f * m * ts) / cnt + m * m;
      mean[b * TLEN + t] = m;
      istd[b * TLEN + t] = rsqrtf(var + 1e-8f);
    }
    cs += ls[255]; cq += lq[255];
    __syncthreads();
  }
}

// normalize + convert to bf16 in (B, TPAD, CPAD) layout (feeds WMMA B-fragments)
__global__ __launch_bounds__(256)
void apply_bf16t_kernel(const float* __restrict__ a, const float* __restrict__ g,
                        const float* __restrict__ bb, const float* __restrict__ mean,
                        const float* __restrict__ istd, __bf16* __restrict__ o,
                        int C, int CPAD) {
  int t = blockIdx.x * 256 + threadIdx.x;
  int c = blockIdx.y, b = blockIdx.z;
  if (t >= TLEN) return;
  float v = 0.f;
  if (c < C) {
    long idx = ((long)b * C + c) * TLEN + t;
    v = (a[idx] - mean[b * TLEN + t]) * istd[b * TLEN + t] * g[c] + bb[c];
  }
  o[((long)b * TPAD + t) * CPAD + c] = (__bf16)v;
}

// fused: cln1-normalize on read  ->  depthwise causal dilated conv (K=3)
//        -> + bias -> PReLU.  Removes a full 66MB elementwise pass.
__global__ __launch_bounds__(256)
void dwconv_cln_kernel(const float* __restrict__ x,      // prelu'd conv1 out (B,H,T)
                       const float* __restrict__ g, const float* __restrict__ bb,
                       const float* __restrict__ mean, const float* __restrict__ istd,
                       const float* __restrict__ w, const float* __restrict__ bias,
                       const float* __restrict__ alpha_p,
                       float* __restrict__ y, int dil) {
  int t = blockIdx.x * 256 + threadIdx.x;
  int c = blockIdx.y, b = blockIdx.z;
  if (t >= TLEN) return;
  const float* xp = x + ((long)b * HID + c) * TLEN;
  const float* mp = mean + b * TLEN;
  const float* ip = istd + b * TLEN;
  float gc = g[c], bc = bb[c];
  float acc = bias[c];
#pragma unroll
  for (int j = 0; j < 3; ++j) {
    int tt = t - (2 - j) * dil;
    if (tt >= 0) {
      float xv = (xp[tt] - mp[tt]) * ip[tt] * gc + bc;   // cln1 applied per tap
      acc += w[c * 3 + j] * xv;
    }
  }
  float al = *alpha_p;
  if (acc < 0.f) acc *= al;
  y[((long)b * HID + c) * TLEN + t] = acc;
}

// ---------------- bf16 WMMA GEMM (single weight matrix) ---------------------
// Y(b, m, n) = sum_k W[m,k] * X[b, n, k] + bias[m]   (n = time)
__global__ __launch_bounds__(128)
void gemm_wmma_kernel(const __bf16* __restrict__ W, int WR,       // padded weight rows
                      const float* __restrict__ bias,
                      const __bf16* __restrict__ X, int CI,       // K (mult of 32)
                      float* __restrict__ Yf, int CO,             // real output rows
                      __bf16* __restrict__ Ybf,                   // optional (B,TPAD,CO)
                      const float* __restrict__ prelu_p) {
  int wave = threadIdx.x >> 5;
  int lane = threadIdx.x & 31;
  int half = lane >> 4;
  int l16  = lane & 15;
  int m0 = blockIdx.x * 64 + wave * 16;
  int n0 = blockIdx.y * 64;
  int b  = blockIdx.z;

  int mrow = m0 + l16; if (mrow > WR - 1) mrow = WR - 1;   // clamp for padded M
  const __bf16* Wrow  = W + (long)mrow * CI;
  const __bf16* Xbase = X + (long)b * TPAD * CI;

  v8f acc[4] = {};
  for (int k0 = 0; k0 < CI; k0 += 32) {
    v8bf alo = *(const v8bf*)(Wrow + k0 + half * 16);
    v8bf ahi = *(const v8bf*)(Wrow + k0 + half * 16 + 8);
    v16bf a;
#pragma unroll
    for (int e = 0; e < 8; ++e) { a[e] = alo[e]; a[8 + e] = ahi[e]; }
#pragma unroll
    for (int j = 0; j < 4; ++j) {
      const __bf16* Xrow = Xbase + (long)(n0 + j * 16 + l16) * CI + k0 + half * 16;
      v16bf bf = *(const v16bf*)Xrow;
      acc[j] = __builtin_amdgcn_wmma_f32_16x16x32_bf16(
          false, a, false, bf, (short)0, acc[j], false, false);
    }
  }

  float alpha = prelu_p ? *prelu_p : 0.f;
#pragma unroll
  for (int j = 0; j < 4; ++j) {
    int n = n0 + j * 16 + l16;
    if (n >= TLEN) continue;
#pragma unroll
    for (int r = 0; r < 8; ++r) {
      int m = m0 + r + 8 * half;
      if (m >= CO) continue;
      float v = acc[j][r] + bias[m];
      if (prelu_p && v < 0.f) v *= alpha;
      long yi = ((long)b * CO + m) * TLEN + n;
      Yf[yi] = v;
      if (Ybf) Ybf[((long)b * TPAD + n) * CO + m] = (__bf16)v;
    }
  }
}

// ---------------- dual bf16 WMMA GEMM (res + skip share X fragments) --------
// Y1 += W1*X + b1 ; Y2 += W2*X + b2 ; B-fragments loaded once, 8 WMMAs/K-step.
__global__ __launch_bounds__(128)
void gemm_dual_wmma_kernel(const __bf16* __restrict__ W1, const __bf16* __restrict__ W2,
                           const float* __restrict__ bias1, const float* __restrict__ bias2,
                           const __bf16* __restrict__ X, int CI,
                           float* __restrict__ Y1, float* __restrict__ Y2, int CO,
                           __bf16* __restrict__ Ybf1, __bf16* __restrict__ Ybf2) {
  int wave = threadIdx.x >> 5;
  int lane = threadIdx.x & 31;
  int half = lane >> 4;
  int l16  = lane & 15;
  int m0 = blockIdx.x * 64 + wave * 16;
  int n0 = blockIdx.y * 64;
  int b  = blockIdx.z;

  const __bf16* W1row = W1 + (long)(m0 + l16) * CI;
  const __bf16* W2row = W2 + (long)(m0 + l16) * CI;
  const __bf16* Xbase = X + (long)b * TPAD * CI;

  v8f acc1[4] = {};
  v8f acc2[4] = {};
  for (int k0 = 0; k0 < CI; k0 += 32) {
    v8bf a1lo = *(const v8bf*)(W1row + k0 + half * 16);
    v8bf a1hi = *(const v8bf*)(W1row + k0 + half * 16 + 8);
    v8bf a2lo = *(const v8bf*)(W2row + k0 + half * 16);
    v8bf a2hi = *(const v8bf*)(W2row + k0 + half * 16 + 8);
    v16bf a1, a2;
#pragma unroll
    for (int e = 0; e < 8; ++e) {
      a1[e] = a1lo[e]; a1[8 + e] = a1hi[e];
      a2[e] = a2lo[e]; a2[8 + e] = a2hi[e];
    }
#pragma unroll
    for (int j = 0; j < 4; ++j) {
      const __bf16* Xrow = Xbase + (long)(n0 + j * 16 + l16) * CI + k0 + half * 16;
      v16bf bf = *(const v16bf*)Xrow;
      acc1[j] = __builtin_amdgcn_wmma_f32_16x16x32_bf16(
          false, a1, false, bf, (short)0, acc1[j], false, false);
      acc2[j] = __builtin_amdgcn_wmma_f32_16x16x32_bf16(
          false, a2, false, bf, (short)0, acc2[j], false, false);
    }
  }

#pragma unroll
  for (int j = 0; j < 4; ++j) {
    int n = n0 + j * 16 + l16;
    if (n >= TLEN) continue;
#pragma unroll
    for (int r = 0; r < 8; ++r) {
      int m = m0 + r + 8 * half;
      long yi = ((long)b * CO + m) * TLEN + n;
      float v1 = Y1[yi] + acc1[j][r] + bias1[m];
      float v2 = Y2[yi] + acc2[j][r] + bias2[m];
      Y1[yi] = v1;
      Y2[yi] = v2;
      long ti = ((long)b * TPAD + n) * CO + m;
      if (Ybf1) Ybf1[ti] = (__bf16)v1;
      if (Ybf2) Ybf2[ti] = (__bf16)v2;
    }
  }
}

// ---------------- host orchestration ----------------------------------------
extern "C" void kernel_launch(void* const* d_in, const int* in_sizes, int n_in,
                              void* d_out, int out_size, void* d_ws, size_t ws_size,
                              hipStream_t stream) {
  (void)in_sizes; (void)n_in; (void)out_size; (void)ws_size;
  const float* x       = (const float*)d_in[0];
  const float* ln_g    = (const float*)d_in[1];
  const float* ln_b    = (const float*)d_in[2];
  const float* bn_w    = (const float*)d_in[3];
  const float* bn_b    = (const float*)d_in[4];
  const float* conv1_w = (const float*)d_in[5];
  const float* conv1_b = (const float*)d_in[6];
  const float* dconv_w = (const float*)d_in[7];
  const float* dconv_b = (const float*)d_in[8];
  const float* cln1_g  = (const float*)d_in[9];
  const float* cln1_b  = (const float*)d_in[10];
  const float* cln2_g  = (const float*)d_in[11];
  const float* cln2_b  = (const float*)d_in[12];
  const float* res_w   = (const float*)d_in[13];
  const float* res_b   = (const float*)d_in[14];
  const float* skip_w  = (const float*)d_in[15];
  const float* skip_b  = (const float*)d_in[16];
  const float* prelu1  = (const float*)d_in[17];
  const float* prelu2  = (const float*)d_in[18];
  const float* out_w   = (const float*)d_in[19];
  const float* out_b   = (const float*)d_in[20];

  char* ws = (char*)d_ws;
  size_t off = 0;
  auto carve = [&](size_t bytes) -> char* {
    char* p = ws + off; off += (bytes + 1023) & ~(size_t)1023; return p;
  };
  __bf16* wbn    = (__bf16*)carve((size_t)BND * CINP * 2);
  __bf16* wc1    = (__bf16*)carve((size_t)NBLK * HID * BND * 2);
  __bf16* wres   = (__bf16*)carve((size_t)NBLK * BND * HID * 2);
  __bf16* wskip  = (__bf16*)carve((size_t)NBLK * BND * HID * 2);
  __bf16* wout   = (__bf16*)carve((size_t)COUTP * BND * 2);
  __bf16* xnbf   = (__bf16*)carve((size_t)BATCH * TPAD * CINP * 2);
  float*  outf   = (float*) carve((size_t)BATCH * BND * TLEN * 4);
  __bf16* outbf  = (__bf16*)carve((size_t)BATCH * TPAD * BND * 2);
  float*  skipf  = (float*) carve((size_t)BATCH * BND * TLEN * 4);
  __bf16* skipbf = (__bf16*)carve((size_t)BATCH * TPAD * BND * 2);
  float*  hf     = (float*) carve((size_t)BATCH * HID * TLEN * 4);
  float*  h2f    = (float*) carve((size_t)BATCH * HID * TLEN * 4);
  __bf16* h3bf   = (__bf16*)carve((size_t)BATCH * TPAD * HID * 2);
  float*  ssum   = (float*) carve((size_t)BATCH * TLEN * 4);
  float*  sqv    = (float*) carve((size_t)BATCH * TLEN * 4);
  float*  mn     = (float*) carve((size_t)BATCH * TLEN * 4);
  float*  isd    = (float*) carve((size_t)BATCH * TLEN * 4);

  auto cvt = [&](const float* s, __bf16* d, int Rd, int Cd, int Rs, int Cs) {
    long n = (long)Rd * Cd;
    convert_pad_kernel<<<(unsigned)((n + 255) / 256), 256, 0, stream>>>(s, d, Rd, Cd, Rs, Cs);
  };
  cvt(bn_w,    wbn,   BND,        CINP, BND,        CIN);
  cvt(conv1_w, wc1,   NBLK * HID, BND,  NBLK * HID, BND);
  cvt(res_w,   wres,  NBLK * BND, HID,  NBLK * BND, HID);
  cvt(skip_w,  wskip, NBLK * BND, HID,  NBLK * BND, HID);
  cvt(out_w,   wout,  COUTP,      BND,  CIN,        BND);

  { long n = (long)BATCH * BND * TLEN;
    zero_f32_kernel<<<(unsigned)((n + 255) / 256), 256, 0, stream>>>(skipf, n); }

  dim3 gstat((TLEN + 255) / 256, BATCH);

  // input cumulative LN -> bf16 (B,TPAD,288)
  colstats_kernel<<<gstat, 256, 0, stream>>>(x, ssum, sqv, CIN);
  scan_kernel<<<BATCH, 256, 0, stream>>>(ssum, sqv, mn, isd, CIN);
  apply_bf16t_kernel<<<dim3((TLEN + 255) / 256, CINP, BATCH), 256, 0, stream>>>(
      x, ln_g, ln_b, mn, isd, xnbf, CIN, CINP);

  // bottleneck: out = bn_w * xn + bn_b
  gemm_wmma_kernel<<<dim3(BND / 64, (TLEN + 63) / 64, BATCH), 128, 0, stream>>>(
      wbn, BND, bn_b, xnbf, CINP, outf, BND, outbf, nullptr);

  for (int i = 0; i < NBLK; ++i) {
    int dil = 1 << (i % 8);
    // conv1 + prelu1 -> hf (B,H,T)
    gemm_wmma_kernel<<<dim3(HID / 64, (TLEN + 63) / 64, BATCH), 128, 0, stream>>>(
        wc1 + (size_t)i * HID * BND, HID, conv1_b + i * HID,
        outbf, BND, hf, HID, nullptr, prelu1 + i);
    // cln1 stats (of prelu'd hf)
    colstats_kernel<<<gstat, 256, 0, stream>>>(hf, ssum, sqv, HID);
    scan_kernel<<<BATCH, 256, 0, stream>>>(ssum, sqv, mn, isd, HID);
    // fused cln1-normalize + depthwise causal conv + bias + prelu2 -> h2f
    dwconv_cln_kernel<<<dim3((TLEN + 255) / 256, HID, BATCH), 256, 0, stream>>>(
        hf, cln1_g + i * HID, cln1_b + i * HID, mn, isd,
        dconv_w + (size_t)i * HID * 3, dconv_b + i * HID, prelu2 + i, h2f, dil);
    // cln2 -> bf16 (B,TPAD,512)
    colstats_kernel<<<gstat, 256, 0, stream>>>(h2f, ssum, sqv, HID);
    scan_kernel<<<BATCH, 256, 0, stream>>>(ssum, sqv, mn, isd, HID);
    apply_bf16t_kernel<<<dim3((TLEN + 255) / 256, HID, BATCH), 256, 0, stream>>>(
        h2f, cln2_g + i * HID, cln2_b + i * HID, mn, isd, h3bf, HID, HID);
    // fused residual + skip GEMM (shared B-fragments):
    //   out += res_w*h3 + res_b  (refresh bf16 out copy)
    //   skip += skip_w*h3 + skip_b (bf16 copy only on last block)
    gemm_dual_wmma_kernel<<<dim3(BND / 64, (TLEN + 63) / 64, BATCH), 128, 0, stream>>>(
        wres + (size_t)i * BND * HID, wskip + (size_t)i * BND * HID,
        res_b + i * BND, skip_b + i * BND,
        h3bf, HID, outf, skipf, BND,
        outbf, (i == NBLK - 1) ? skipbf : nullptr);
  }

  // output projection: d_out = out_w * skip + out_b  (M padded 257 -> 272)
  gemm_wmma_kernel<<<dim3((CIN + 63) / 64, (TLEN + 63) / 64, BATCH), 128, 0, stream>>>(
      wout, COUTP, out_b, skipbf, BND, (float*)d_out, CIN, nullptr, nullptr);
}
